// MemoryFusion_73400991089000
// MI455X (gfx1250) — matrix operand order repaired and checked by reading, not compiled, not run on hardware
//
#include <hip/hip_runtime.h>
#include <hip/hip_bf16.h>

#define D_MODEL 2048
#define NUM_HEADS 16
#define D_K 128
#define BATCH 4
#define SEQ 2048
#define CHUNK 1024

typedef __attribute__((ext_vector_type(16))) __bf16 v16bf;
typedef __attribute__((ext_vector_type(8)))  __bf16 v8bf;
typedef __attribute__((ext_vector_type(8)))  float  v8f;

// ---------------------------------------------------------------------------
// WMMA fragment loaders (wave32, v_wmma_f32_16x16x32_bf16)
//
// A matrix (16 M x 32 K), row-major source with leading dim `ld`:
//   lane l<16 : row M=l,    K = {0..7} in elems 0..7, {16..23} in elems 8..15
//   lane l>=16: row M=l-16, K = {8..15} and {24..31}
__device__ __forceinline__ v16bf load_frag_a(const __bf16* base, int ld) {
  const int lane = threadIdx.x & 31;
  const int row  = lane & 15;
  const int koff = (lane >> 4) << 3;            // 0 or 8
  const __bf16* p = base + (size_t)row * ld + koff;
  v16bf a;
  ((v8bf*)&a)[0] = *(const v8bf*)(p);
  ((v8bf*)&a)[1] = *(const v8bf*)(p + 16);
  return a;
}

// B matrix (32 K x 16 N) where memory holds B^T row-major (rows = N, cols = K):
//   lane l<16 : col N=l,    K = {0..7} in elems 0..7, {16..23} in elems 8..15
//   lane l>=16: col N=l-16, K = {8..15} and {24..31}
__device__ __forceinline__ v16bf load_frag_b_t(const __bf16* base, int ld) {
  const int lane = threadIdx.x & 31;
  const int n  = lane & 15;
  const int kb = (lane >> 4) << 3;              // 0 or 8
  const __bf16* p = base + (size_t)n * ld + kb;
  v16bf b;
  ((v8bf*)&b)[0] = *(const v8bf*)(p);
  ((v8bf*)&b)[1] = *(const v8bf*)(p + 16);
  return b;
}

// ---------------------------------------------------------------------------
// Kernel 0: transpose the three projection weights once (Wt[e][d] = W[d][e]).
// 64x64 bf16 tiles staged global->LDS with the CDNA5 async data mover
// (global_load_async_to_lds_b128 + s_wait_asynccnt), then written back
// coalesced. Cost: ~50 MB total traffic (~2us at 23.3 TB/s) -- noise vs the
// 206 GFLOP projection GEMM it unblocks.
__global__ __launch_bounds__(256)
void wtrans_kernel(const __bf16* __restrict__ Wq, const __bf16* __restrict__ Wk,
                   const __bf16* __restrict__ Wv,
                   __bf16* __restrict__ Wqt, __bf16* __restrict__ Wkt,
                   __bf16* __restrict__ Wvt) {
  const __bf16* W  = (blockIdx.z == 0) ? Wq  : (blockIdx.z == 1) ? Wk  : Wv;
  __bf16*       Wt = (blockIdx.z == 0) ? Wqt : (blockIdx.z == 1) ? Wkt : Wvt;

  __shared__ __align__(16) __bf16 tile[64][72];   // 72*2B = 144B row stride (16B multiple)

  const int t  = threadIdx.x;
  const int d0 = blockIdx.y * 64;                 // source row block (d)
  const int e0 = blockIdx.x * 64;                 // source col block (e)

  // Stage 64x64 tile: 512 chunks of 16B; 2 async copies per thread.
#pragma unroll
  for (int i = 0; i < 2; ++i) {
    const int c    = t + i * 256;
    const int r    = c >> 3;                      // 0..63
    const int ccol = (c & 7) * 8;                 // 0,8,..,56
    const __bf16* g = W + (size_t)(d0 + r) * D_MODEL + e0 + ccol;
    const unsigned lds_off = (unsigned)(size_t)&tile[r][ccol];
    asm volatile("global_load_async_to_lds_b128 %0, %1, off"
                 :: "v"(lds_off), "v"(g) : "memory");
  }
  asm volatile("s_wait_asynccnt 0" ::: "memory");
  __syncthreads();

  // Write transposed, coalesced 16B chunks: Wt[e0+r][d0+ccol..+8]
#pragma unroll
  for (int i = 0; i < 2; ++i) {
    const int c    = t + i * 256;
    const int r    = c >> 3;
    const int ccol = (c & 7) * 8;
    v8bf val;
    __bf16* ev = (__bf16*)&val;
#pragma unroll
    for (int j = 0; j < 8; ++j) ev[j] = tile[ccol + j][r];
    *(v8bf*)(Wt + (size_t)(e0 + r) * D_MODEL + d0 + ccol) = val;
  }
}

// ---------------------------------------------------------------------------
// Kernel 1: fused QKV projection.  C = X @ W + bias, M=B*S=8192, N=K=2048.
// Uses pre-transposed weights so the B operand loads as 2x global_load_b128.
// Each wave owns a 16x128 strip: one A fragment feeds 8 WMMAs.
// blockIdx.y selects {Q,K,V}. V output is stored transposed: Vt[b, e, s].
__global__ __launch_bounds__(128)
void qkv_proj_kernel(const __bf16* __restrict__ X,
                     const __bf16* __restrict__ Wqt, const __bf16* __restrict__ bq,
                     const __bf16* __restrict__ Wkt, const __bf16* __restrict__ bk,
                     const __bf16* __restrict__ Wvt, const __bf16* __restrict__ bvp,
                     __bf16* __restrict__ Qo, __bf16* __restrict__ Ko,
                     __bf16* __restrict__ Vt) {
  const int mode = blockIdx.y;                               // 0=Q 1=K 2=V
  const __bf16* Wt   = (mode == 0) ? Wqt : (mode == 1) ? Wkt : Wvt;
  const __bf16* bias = (mode == 0) ? bq  : (mode == 1) ? bk  : bvp;

  const int wave = threadIdx.x >> 5;
  const int lane = threadIdx.x & 31;
  const int strip = blockIdx.x * 4 + wave;                   // 8192 strips
  const int m0 = (strip >> 4) << 4;                          // M tile base (512 tiles)
  const int n0 = (strip & 15) << 7;                          // N strip base (128 wide)

  v8f acc[8] = {};
  for (int k = 0; k < D_MODEL; k += 32) {
    const __bf16* abase = X + (size_t)m0 * D_MODEL + k;
    __builtin_prefetch(abase + 256, 0, 1);                   // global_prefetch_b8
    v16bf a = load_frag_a(abase, D_MODEL);
#pragma unroll
    for (int j = 0; j < 8; ++j) {
      const __bf16* bbase = Wt + (size_t)(n0 + j * 16) * D_MODEL + k;
      v16bf bf = load_frag_b_t(bbase, D_MODEL);
      acc[j] = __builtin_amdgcn_wmma_f32_16x16x32_bf16(
          false, a, false, bf, (short)0, acc[j], false, false);
    }
  }

  const int col   = lane & 15;
  const int rbase = (lane >> 4) << 3;
#pragma unroll
  for (int j = 0; j < 8; ++j) {
    const int e = n0 + j * 16 + col;
    const float bval = (float)bias[e];
#pragma unroll
    for (int v = 0; v < 8; ++v) {
      const int m = m0 + v + rbase;
      const float val = acc[j][v] + bval;
      if (mode == 2) {
        const int bb = m >> 11;                              // m / SEQ
        const int s  = m & (SEQ - 1);
        Vt[((size_t)bb * D_MODEL + e) * SEQ + s] = (__bf16)val;
      } else {
        __bf16* out = (mode == 0) ? Qo : Ko;
        out[(size_t)m * D_MODEL + e] = (__bf16)val;
      }
    }
  }
}

// ---------------------------------------------------------------------------
// Kernel 2: flash attention. One workgroup (8 waves) = 16 query rows of (b,h).
// Keys processed in two 1024-chunks with online softmax; scores/probs in LDS
// as bf16 (32KB), stats in f32.
__global__ __launch_bounds__(256)
void attn_kernel(const __bf16* __restrict__ Q, const __bf16* __restrict__ K,
                 const __bf16* __restrict__ Vt, __bf16* __restrict__ O) {
  __shared__ __align__(16) __bf16 sP[16 * CHUNK];            // scores -> probs
  __shared__ float sRed[16][17];
  __shared__ float sM[16], sL[16], sSF[16];

  const int b  = blockIdx.z, h = blockIdx.y, q0 = blockIdx.x << 4;
  const int tid  = threadIdx.x;
  const int wave = tid >> 5, lane = tid & 31;
  const int col = lane & 15, rbase = (lane >> 4) << 3;

  const __bf16* Qb = Q  + ((size_t)(b * SEQ + q0)) * D_MODEL + h * D_K;
  const __bf16* Kb = K  + ((size_t)b * SEQ) * D_MODEL + h * D_K;
  const __bf16* Vb = Vt + ((size_t)b * D_MODEL + h * D_K) * SEQ;

  if (tid < 16) { sM[tid] = -1e30f; sL[tid] = 0.0f; }

  // Q fragments are reused across all key tiles and both chunks.
  v16bf aq[4];
#pragma unroll
  for (int kk = 0; kk < 4; ++kk) aq[kk] = load_frag_a(Qb + kk * 32, D_MODEL);

  v8f oacc = {};
  const float scale = 0.08838834764831845f;                  // 1/sqrt(128)
  const int r = tid >> 4, jj = tid & 15;

  __syncthreads();

  for (int c = 0; c < 2; ++c) {
    const int kc = c * CHUNK;

    // ---- Phase 1: scores = scale * Q @ K^T for this chunk ----
#pragma unroll
    for (int t = 0; t < 8; ++t) {
      const int n0 = wave * 128 + t * 16;                    // key offset in chunk
      v8f sacc = {};
#pragma unroll
      for (int kk = 0; kk < 4; ++kk) {
        v16bf bk_ = load_frag_b_t(Kb + (size_t)(kc + n0) * D_MODEL + kk * 32, D_MODEL);
        sacc = __builtin_amdgcn_wmma_f32_16x16x32_bf16(
            false, aq[kk], false, bk_, (short)0, sacc, false, false);
      }
#pragma unroll
      for (int v = 0; v < 8; ++v)
        sP[(v + rbase) * CHUNK + n0 + col] = (__bf16)(sacc[v] * scale);
    }
    __syncthreads();

    // ---- Phase 2a: per-row chunk max, online-softmax stats update ----
    float mx = -1e30f;
#pragma unroll 4
    for (int i = 0; i < CHUNK / 16; ++i)
      mx = fmaxf(mx, (float)sP[r * CHUNK + jj + (i << 4)]);
    sRed[r][jj] = mx;
    __syncthreads();
    if (jj == 0) {
      float cm = sRed[r][0];
      for (int i = 1; i < 16; ++i) cm = fmaxf(cm, sRed[r][i]);
      const float mold = sM[r];
      const float mnew = fmaxf(mold, cm);
      const float sf   = __expf(mold - mnew);
      sM[r] = mnew; sSF[r] = sf; sL[r] *= sf;
    }
    __syncthreads();

    // ---- Phase 2b: rescale O accumulators; exp in place; row-sum partials ----
#pragma unroll
    for (int v = 0; v < 8; ++v) oacc[v] *= sSF[v + rbase];

    const float mr = sM[r];
    float sum = 0.0f;
#pragma unroll 4
    for (int i = 0; i < CHUNK / 16; ++i) {
      const int idx = r * CHUNK + jj + (i << 4);
      const float p = __expf((float)sP[idx] - mr);
      sP[idx] = (__bf16)p;
      sum += p;
    }
    sRed[r][jj] = sum;
    __syncthreads();
    if (jj == 0) {
      float s_ = 0.0f;
      for (int i = 0; i < 16; ++i) s_ += sRed[r][i];
      sL[r] += s_;
    }
    __syncthreads();

    // ---- Phase 3: O += P @ V ; wave owns dk columns [wave*16, wave*16+16) ----
#pragma unroll 4
    for (int kt = 0; kt < CHUNK / 32; ++kt) {
      v16bf ap  = load_frag_a((const __bf16*)sP + kt * 32, CHUNK);
      v16bf bv_ = load_frag_b_t(Vb + (size_t)(wave * 16) * SEQ + kc + kt * 32, SEQ);
      oacc = __builtin_amdgcn_wmma_f32_16x16x32_bf16(
          false, ap, false, bv_, (short)0, oacc, false, false);
    }
    __syncthreads();
  }

  // Normalize by softmax denominator and store [B,S,D]
#pragma unroll
  for (int v = 0; v < 8; ++v) {
    const int row = v + rbase;
    const float inv = 1.0f / sL[row];
    O[((size_t)(b * SEQ + q0 + row)) * D_MODEL + h * D_K + wave * 16 + col] =
        (__bf16)(oacc[v] * inv);
  }
}

// ---------------------------------------------------------------------------
// Kernel 3: Obar[b,d] = mean_s O[b,s,d]   (mean commutes with the out-proj)
__global__ __launch_bounds__(256)
void rowmean_kernel(const __bf16* __restrict__ O, float* __restrict__ Obar) {
  const int t = blockIdx.x * 256 + threadIdx.x;              // 0..8191
  const int b = t >> 11, d = t & (D_MODEL - 1);
  const __bf16* p = O + (size_t)b * SEQ * D_MODEL + d;
  float s = 0.0f;
  for (int i = 0; i < SEQ; ++i) s += (float)p[(size_t)i * D_MODEL];
  Obar[t] = s * (1.0f / SEQ);
}

// Kernel 4: Y[b,e] = Obar[b,:] @ Wo[:,e] + bo[e]   (tiny M=4 GEMM)
__global__ __launch_bounds__(256)
void outproj_kernel(const float* __restrict__ Obar, const __bf16* __restrict__ Wo,
                    const __bf16* __restrict__ bo, __bf16* __restrict__ Y) {
  const int t = blockIdx.x * 256 + threadIdx.x;              // 0..8191
  const int b = t >> 11, e = t & (D_MODEL - 1);
  const float* x = Obar + b * D_MODEL;
  float acc = 0.0f;
  for (int d = 0; d < D_MODEL; ++d)
    acc += x[d] * (float)Wo[(size_t)d * D_MODEL + e];
  Y[t] = (__bf16)(acc + (float)bo[e]);
}

// ---------------------------------------------------------------------------
extern "C" void kernel_launch(void* const* d_in, const int* in_sizes, int n_in,
                              void* d_out, int out_size, void* d_ws, size_t ws_size,
                              hipStream_t stream) {
  (void)in_sizes; (void)n_in; (void)out_size; (void)ws_size;
  const __bf16* X  = (const __bf16*)d_in[0];
  const __bf16* Wq = (const __bf16*)d_in[1];
  const __bf16* bq = (const __bf16*)d_in[2];
  const __bf16* Wk = (const __bf16*)d_in[3];
  const __bf16* bk = (const __bf16*)d_in[4];
  const __bf16* Wv = (const __bf16*)d_in[5];
  const __bf16* bv = (const __bf16*)d_in[6];
  const __bf16* Wo = (const __bf16*)d_in[7];
  const __bf16* bo = (const __bf16*)d_in[8];

  char* ws = (char*)d_ws;
  const size_t MAT  = (size_t)BATCH * SEQ * D_MODEL * sizeof(__bf16);  // 33.5 MB
  const size_t WMAT = (size_t)D_MODEL * D_MODEL * sizeof(__bf16);      // 8.4 MB
  __bf16* Qb   = (__bf16*)(ws + 0 * MAT);
  __bf16* Kb   = (__bf16*)(ws + 1 * MAT);
  __bf16* Vt   = (__bf16*)(ws + 2 * MAT);
  __bf16* O    = (__bf16*)(ws + 3 * MAT);
  float*  Obar = (float*) (ws + 4 * MAT);                              // 32 KB
  char*   wbase = ws + 4 * MAT + (1 << 16);
  __bf16* Wqt  = (__bf16*)(wbase + 0 * WMAT);
  __bf16* Wkt  = (__bf16*)(wbase + 1 * WMAT);
  __bf16* Wvt  = (__bf16*)(wbase + 2 * WMAT);

  wtrans_kernel<<<dim3(D_MODEL / 64, D_MODEL / 64, 3), 256, 0, stream>>>(
      Wq, Wk, Wv, Wqt, Wkt, Wvt);
  qkv_proj_kernel<<<dim3(2048, 3), 128, 0, stream>>>(X, Wqt, bq, Wkt, bk, Wvt, bv,
                                                     Qb, Kb, Vt);
  attn_kernel<<<dim3(SEQ / 16, NUM_HEADS, BATCH), 256, 0, stream>>>(Qb, Kb, Vt, O);
  rowmean_kernel<<<32, 256, 0, stream>>>(O, Obar);
  outproj_kernel<<<32, 256, 0, stream>>>(Obar, Wo, bo, (__bf16*)d_out);
}
